// TransformerEncoderModel_21895743275718
// MI455X (gfx1250) — compile-verified, matrix operand order checked by reading, not run on hardware
//
#include <hip/hip_runtime.h>
#include <math.h>

#define BB 4
#define SS 2048
#define INDIM 512
#define DD 256
#define HH 4
#define LL 4
#define FFD 2048
#define HD 64
#define BS (BB*SS)   /* 8192 */

typedef __attribute__((ext_vector_type(16))) _Float16     v16h;
typedef __attribute__((ext_vector_type(8)))  float        v8f;
typedef __attribute__((ext_vector_type(4)))  unsigned int u32x4;

static __device__ __forceinline__ v8f v8f_zero() {
  v8f z = {0.f, 0.f, 0.f, 0.f, 0.f, 0.f, 0.f, 0.f};
  return z;
}

// Load a 16x32 f16 fragment (A-layout; B symmetric column layout) from LDS.
// ISA: lanes 0-15 rows 0-15 hold K {0..7, 16..23}; lanes 16-31 K {8..15, 24..31}.
static __device__ __forceinline__ v16h frag_ld(const _Float16* base, int ld) {
  const int lane = threadIdx.x & 31;
  const int row  = lane & 15;
  const int half = lane >> 4;
  const _Float16* p = base + row * ld + half * 8;
  union { v16h v; u32x4 q[2]; } f;
  f.q[0] = *(const u32x4*)(p);
  f.q[1] = *(const u32x4*)(p + 16);
  return f.v;
}

static __device__ __forceinline__ v8f wmma16(v16h a, v16h b, v8f c) {
  return __builtin_amdgcn_wmma_f32_16x16x32_f16(false, a, false, b, (short)0, c,
                                                false, false);
}

static __device__ __forceinline__ void cvt_store16(_Float16* dst,
                                                   float4 f0, float4 f1,
                                                   float4 f2, float4 f3) {
  union { _Float16 h[16]; u32x4 q[2]; } t;
  t.h[0]=(_Float16)f0.x;  t.h[1]=(_Float16)f0.y;  t.h[2]=(_Float16)f0.z;  t.h[3]=(_Float16)f0.w;
  t.h[4]=(_Float16)f1.x;  t.h[5]=(_Float16)f1.y;  t.h[6]=(_Float16)f1.z;  t.h[7]=(_Float16)f1.w;
  t.h[8]=(_Float16)f2.x;  t.h[9]=(_Float16)f2.y;  t.h[10]=(_Float16)f2.z; t.h[11]=(_Float16)f2.w;
  t.h[12]=(_Float16)f3.x; t.h[13]=(_Float16)f3.y; t.h[14]=(_Float16)f3.z; t.h[15]=(_Float16)f3.w;
  u32x4* d = (u32x4*)dst;
  d[0] = t.q[0]; d[1] = t.q[1];
}

// ---------------------------------------------------------------------------
// GEMM: C[M,N] = A[M,K] * W[N,K]^T + bias, MODE: 0 = plain, 1 = +PE, 2 = +ReLU
// Block tile 128x128, 256 threads (8 waves), wave tile 64x32 (4x2 WMMA tiles).
// Global loads for step kb+1 are issued before the WMMAs of step kb.
// ---------------------------------------------------------------------------
template <int MODE>
__global__ __launch_bounds__(256)
void gemm_f16w(const float* __restrict__ A, const float* __restrict__ W,
               const float* __restrict__ bias, float* __restrict__ C,
               int N, int K, const float* __restrict__ pe)
{
  __shared__ _Float16 sA[128 * 40];
  __shared__ _Float16 sB[128 * 40];

  const int tid  = threadIdx.x;
  const int lane = tid & 31;
  const int wave = tid >> 5;
  const int wm   = wave >> 2;   // 0..1
  const int wn   = wave & 3;    // 0..3
  const int rowBase = blockIdx.y * 128;
  const int colBase = blockIdx.x * 128;

  v8f acc[4][2];
#pragma unroll
  for (int mi = 0; mi < 4; ++mi)
#pragma unroll
    for (int ni = 0; ni < 2; ++ni) acc[mi][ni] = v8f_zero();

  const int r  = tid >> 1;          // 0..127 staging row
  const int kh = (tid & 1) << 4;    // 0 or 16

  const float* gA = A + (size_t)(rowBase + r) * K + kh;
  const float* gW = W + (size_t)(colBase + r) * K + kh;
  _Float16* dA = &sA[r * 40 + kh];
  _Float16* dW = &sB[r * 40 + kh];

  float4 a0, a1, a2, a3, w0, w1, w2, w3;
  a0 = *(const float4*)(gA + 0);  a1 = *(const float4*)(gA + 4);
  a2 = *(const float4*)(gA + 8);  a3 = *(const float4*)(gA + 12);
  w0 = *(const float4*)(gW + 0);  w1 = *(const float4*)(gW + 4);
  w2 = *(const float4*)(gW + 8);  w3 = *(const float4*)(gW + 12);

  const int nK = K >> 5;
  for (int kb = 0; kb < nK; ++kb) {
    cvt_store16(dA, a0, a1, a2, a3);
    cvt_store16(dW, w0, w1, w2, w3);
    __syncthreads();

    if (kb + 1 < nK) {   // prefetch next K-tile while WMMAs run
      const float* pA = gA + (kb + 1) * 32;
      const float* pW = gW + (kb + 1) * 32;
      a0 = *(const float4*)(pA + 0);  a1 = *(const float4*)(pA + 4);
      a2 = *(const float4*)(pA + 8);  a3 = *(const float4*)(pA + 12);
      w0 = *(const float4*)(pW + 0);  w1 = *(const float4*)(pW + 4);
      w2 = *(const float4*)(pW + 8);  w3 = *(const float4*)(pW + 12);
    }

    v16h aF[4], bF[2];
#pragma unroll
    for (int mi = 0; mi < 4; ++mi)
      aF[mi] = frag_ld(&sA[(wm * 64 + mi * 16) * 40], 40);
#pragma unroll
    for (int ni = 0; ni < 2; ++ni)
      bF[ni] = frag_ld(&sB[(wn * 32 + ni * 16) * 40], 40);
#pragma unroll
    for (int mi = 0; mi < 4; ++mi)
#pragma unroll
      for (int ni = 0; ni < 2; ++ni)
        acc[mi][ni] = wmma16(aF[mi], bF[ni], acc[mi][ni]);

    __syncthreads();
  }

  // Epilogue: C layout -> element (row i + 8*half, col lane&15) in VGPR i.
  // All index math 32-bit; row stride handled as i*N offsets off a tile base.
  const int col  = lane & 15;
  const int hsel = lane >> 4;
#pragma unroll
  for (int mi = 0; mi < 4; ++mi) {
#pragma unroll
    for (int ni = 0; ni < 2; ++ni) {
      const int gc  = colBase + wn * 32 + ni * 16 + col;
      const int gr0 = rowBase + wm * 64 + mi * 16 + hsel * 8;
      const float bcol = bias[gc];
      float* cp = C + (gr0 * N + gc);
      const float* pp = (MODE == 1) ? (pe + ((gr0 & (SS - 1)) * N + gc)) : nullptr;
#pragma unroll
      for (int i = 0; i < 8; ++i) {
        float v = acc[mi][ni][i] + bcol;
        if (MODE == 1) v += pp[i * N];
        if (MODE == 2) v = fmaxf(v, 0.f);
        cp[i * N] = v;
      }
    }
  }
}

// ---------------------------------------------------------------------------
// Flash attention: one block = (b,h) x 64-query tile; 4 waves x 16 queries.
// qkv: [B*S, 3*D]; q at col h*64, k at col 256+h*64, v at col 512+h*64.
// ctx: [B*S, D] with head h occupying cols h*64..h*64+63.
// ---------------------------------------------------------------------------
__global__ __launch_bounds__(128)
void attn_kernel(const float* __restrict__ qkv, float* __restrict__ ctx)
{
  __shared__ _Float16 sQ[64 * 72];
  __shared__ _Float16 sK[64 * 72];
  __shared__ _Float16 sV[64 * 72];   // transposed: [d][key]
  __shared__ _Float16 sP[64 * 72];

  const int tid  = threadIdx.x;
  const int lane = tid & 31;
  const int wave = tid >> 5;          // 0..3
  const int qt = blockIdx.x;          // 0..S/64-1
  const int bh = blockIdx.y;          // 0..B*H-1
  const int b = bh >> 2;
  const int h = bh & 3;
  const int rowQ0 = b * SS + qt * 64;

  const int sr   = tid >> 1;          // 0..63
  const int sc32 = (tid & 1) << 5;    // 0 or 32

  { // stage Q (fp32 -> f16)
    const float4* g4 = (const float4*)(qkv + (rowQ0 + sr) * (3 * DD) + h * HD + sc32);
    union { _Float16 hh[32]; u32x4 q[4]; } t;
#pragma unroll
    for (int j = 0; j < 8; ++j) {
      float4 f = g4[j];
      t.hh[4*j+0]=(_Float16)f.x; t.hh[4*j+1]=(_Float16)f.y;
      t.hh[4*j+2]=(_Float16)f.z; t.hh[4*j+3]=(_Float16)f.w;
    }
    u32x4* d = (u32x4*)&sQ[sr * 72 + sc32];
    d[0]=t.q[0]; d[1]=t.q[1]; d[2]=t.q[2]; d[3]=t.q[3];
  }
  __syncthreads();

  const v16h qf0 = frag_ld(&sQ[(wave * 16) * 72 + 0], 72);
  const v16h qf1 = frag_ld(&sQ[(wave * 16) * 72 + 32], 72);

  float run_m[8], run_l[8];
  v8f accO[4];
#pragma unroll
  for (int i = 0; i < 8; ++i) { run_m[i] = -3.0e38f; run_l[i] = 0.f; }
#pragma unroll
  for (int dt = 0; dt < 4; ++dt) accO[dt] = v8f_zero();

  const float scl = 0.125f * 1.44269504088896340736f;  // 1/sqrt(64) * log2(e)

  for (int kt = 0; kt < SS / 64; ++kt) {
    const int rowK0 = b * SS + kt * 64;
    { // stage K
      const float4* g4 = (const float4*)(qkv + (rowK0 + sr) * (3 * DD) + DD + h * HD + sc32);
      union { _Float16 hh[32]; u32x4 q[4]; } t;
#pragma unroll
      for (int j = 0; j < 8; ++j) {
        float4 f = g4[j];
        t.hh[4*j+0]=(_Float16)f.x; t.hh[4*j+1]=(_Float16)f.y;
        t.hh[4*j+2]=(_Float16)f.z; t.hh[4*j+3]=(_Float16)f.w;
      }
      u32x4* d = (u32x4*)&sK[sr * 72 + sc32];
      d[0]=t.q[0]; d[1]=t.q[1]; d[2]=t.q[2]; d[3]=t.q[3];
    }
    { // stage V transposed: thread owns d=sr, keys sc32..sc32+31
      const float* g = qkv + (rowK0 + sc32) * (3 * DD) + 2 * DD + h * HD + sr;
      union { _Float16 hh[32]; u32x4 q[4]; } t;
#pragma unroll
      for (int j = 0; j < 32; ++j) t.hh[j] = (_Float16)g[j * (3 * DD)];
      u32x4* d = (u32x4*)&sV[sr * 72 + sc32];
      d[0]=t.q[0]; d[1]=t.q[1]; d[2]=t.q[2]; d[3]=t.q[3];
    }
    __syncthreads();

    // S = Q K^T : 16 queries x 64 keys per wave
    v8f sAcc[4];
#pragma unroll
    for (int nt = 0; nt < 4; ++nt) sAcc[nt] = v8f_zero();
#pragma unroll
    for (int kk = 0; kk < 2; ++kk) {
      const v16h aq = kk ? qf1 : qf0;
#pragma unroll
      for (int nt = 0; nt < 4; ++nt) {
        v16h bk = frag_ld(&sK[(nt * 16) * 72 + kk * 32], 72);
        sAcc[nt] = wmma16(aq, bk, sAcc[nt]);
      }
    }

    // online softmax: rows live across 16-lane halves; reduce via shfl_xor 1..8
#pragma unroll
    for (int i = 0; i < 8; ++i) {
      float m = fmaxf(fmaxf(sAcc[0][i], sAcc[1][i]), fmaxf(sAcc[2][i], sAcc[3][i]));
      m = fmaxf(m, __shfl_xor(m, 1));
      m = fmaxf(m, __shfl_xor(m, 2));
      m = fmaxf(m, __shfl_xor(m, 4));
      m = fmaxf(m, __shfl_xor(m, 8));
      m *= scl;
      const float nm = fmaxf(run_m[i], m);
      const float f = __builtin_exp2f(run_m[i] - nm);
      run_m[i] = nm;
      run_l[i] *= f;
#pragma unroll
      for (int dt = 0; dt < 4; ++dt) accO[dt][i] = accO[dt][i] * f;
      float rs = 0.f;
#pragma unroll
      for (int nt = 0; nt < 4; ++nt) {
        float p = __builtin_exp2f(sAcc[nt][i] * scl - nm);
        sAcc[nt][i] = p;
        rs += p;
      }
      rs += __shfl_xor(rs, 1);
      rs += __shfl_xor(rs, 2);
      rs += __shfl_xor(rs, 4);
      rs += __shfl_xor(rs, 8);
      run_l[i] += rs;
    }

    // P -> f16 in LDS (wave-private rows; same-wave DS ops stay in order)
    {
      const int col = lane & 15;
      const int hs  = lane >> 4;
#pragma unroll
      for (int nt = 0; nt < 4; ++nt)
#pragma unroll
        for (int i = 0; i < 8; ++i)
          sP[(wave * 16 + i + 8 * hs) * 72 + nt * 16 + col] = (_Float16)sAcc[nt][i];
    }

    // O += P V
#pragma unroll
    for (int kk = 0; kk < 2; ++kk) {
      v16h ap = frag_ld(&sP[(wave * 16) * 72 + kk * 32], 72);
#pragma unroll
      for (int dt = 0; dt < 4; ++dt) {
        v16h bv = frag_ld(&sV[(dt * 16) * 72 + kk * 32], 72);
        accO[dt] = wmma16(ap, bv, accO[dt]);
      }
    }
    __syncthreads();
  }

  // epilogue: ctx = O / l
  const int col = lane & 15;
  const int hs  = lane >> 4;
#pragma unroll
  for (int dt = 0; dt < 4; ++dt) {
#pragma unroll
    for (int i = 0; i < 8; ++i) {
      const int rr = qt * 64 + wave * 16 + i + 8 * hs;
      const int cc = h * HD + dt * 16 + col;
      ctx[(b * SS + rr) * DD + cc] = accO[dt][i] / run_l[i];
    }
  }
}

// ---------------------------------------------------------------------------
// out = LayerNorm(x + y) * w + b      (one wave per 256-elem row)
// ---------------------------------------------------------------------------
__global__ __launch_bounds__(256)
void add_ln_kernel(const float* __restrict__ x, const float* __restrict__ y,
                   const float* __restrict__ w, const float* __restrict__ bsh,
                   float* __restrict__ out)
{
  const int lane = threadIdx.x & 31;
  const int wave = threadIdx.x >> 5;
  const int row  = blockIdx.x * 8 + wave;
  const float* px = x + row * DD;
  const float* py = y + row * DD;
  float v[8];
  float s = 0.f, s2 = 0.f;
#pragma unroll
  for (int j = 0; j < 8; ++j) {
    const int c = lane + j * 32;
    const float t = px[c] + py[c];
    v[j] = t; s += t; s2 += t * t;
  }
#pragma unroll
  for (int o = 1; o < 32; o <<= 1) {
    s  += __shfl_xor(s, o);
    s2 += __shfl_xor(s2, o);
  }
  const float mu  = s * (1.f / DD);
  const float var = s2 * (1.f / DD) - mu * mu;
  const float rs  = rsqrtf(var + 1e-5f);
  float* po = out + row * DD;
#pragma unroll
  for (int j = 0; j < 8; ++j) {
    const int c = lane + j * 32;
    po[c] = (v[j] - mu) * rs * w[c] + bsh[c];
  }
}

// ---------------------------------------------------------------------------
__global__ void pe_kernel(float* __restrict__ pe)
{
  const int idx = blockIdx.x * 256 + threadIdx.x;
  if (idx >= SS * DD) return;
  const int s = idx / DD;
  const int d = idx - s * DD;
  const float div = expf((float)(d & ~1) * (-9.210340371976184f / (float)DD));
  const float ang = (float)s * div;
  pe[idx] = (d & 1) ? cosf(ang) : sinf(ang);
}

// final: out[b,o] = h[b, S-1, :] . W_fc[o,:] + b_fc[o]
__global__ void final_kernel(const float* __restrict__ h, const float* __restrict__ Wfc,
                             const float* __restrict__ bfc, float* __restrict__ out)
{
  const int idx = blockIdx.x * 256 + threadIdx.x;    // 0..2047
  const int bb = idx >> 9;
  const int o  = idx & 511;
  const float* hr = h + (bb * SS + (SS - 1)) * DD;
  const float* wr = Wfc + o * DD;
  float acc = bfc[o];
  for (int d = 0; d < DD; ++d) acc += hr[d] * wr[d];
  out[idx] = acc;
}

// ---------------------------------------------------------------------------
extern "C" void kernel_launch(void* const* d_in, const int* in_sizes, int n_in,
                              void* d_out, int out_size, void* d_ws, size_t ws_size,
                              hipStream_t stream)
{
  const float* x     = (const float*)d_in[0];
  const float* W_in  = (const float*)d_in[1];
  const float* b_in  = (const float*)d_in[2];
  const float* qkv_w = (const float*)d_in[3];
  const float* qkv_b = (const float*)d_in[4];
  const float* out_w = (const float*)d_in[5];
  const float* out_b = (const float*)d_in[6];
  const float* ln1_w = (const float*)d_in[7];
  const float* ln1_b = (const float*)d_in[8];
  const float* ff1_w = (const float*)d_in[9];
  const float* ff1_b = (const float*)d_in[10];
  const float* ff2_w = (const float*)d_in[11];
  const float* ff2_b = (const float*)d_in[12];
  const float* ln2_w = (const float*)d_in[13];
  const float* ln2_b = (const float*)d_in[14];
  const float* W_fc  = (const float*)d_in[15];
  const float* b_fc  = (const float*)d_in[16];
  float* out = (float*)d_out;

  char* ws = (char*)d_ws;
  size_t off = 0;
  float* pe  = (float*)(ws + off); off += (size_t)SS * DD * 4;     // 2 MB
  float* h   = (float*)(ws + off); off += (size_t)BS * DD * 4;     // 8 MB
  float* big = (float*)(ws + off); off += (size_t)BS * FFD * 4;    // 64 MB (qkv & ff reuse)
  float* ctx = (float*)(ws + off); off += (size_t)BS * DD * 4;     // 8 MB
  float* sa  = (float*)(ws + off);                                 // 8 MB
  (void)ws_size; (void)in_sizes; (void)n_in; (void)out_size;

  pe_kernel<<<(SS * DD + 255) / 256, 256, 0, stream>>>(pe);

  // h = x @ W_in^T + b_in + PE
  gemm_f16w<1><<<dim3(DD / 128, BS / 128), 256, 0, stream>>>(
      x, W_in, b_in, h, DD, INDIM, pe);

  for (int l = 0; l < LL; ++l) {
    gemm_f16w<0><<<dim3(3 * DD / 128, BS / 128), 256, 0, stream>>>(
        h, qkv_w + (size_t)l * 3 * DD * DD, qkv_b + (size_t)l * 3 * DD,
        big, 3 * DD, DD, nullptr);

    attn_kernel<<<dim3(SS / 64, BB * HH), 128, 0, stream>>>(big, ctx);

    gemm_f16w<0><<<dim3(DD / 128, BS / 128), 256, 0, stream>>>(
        ctx, out_w + (size_t)l * DD * DD, out_b + (size_t)l * DD,
        sa, DD, DD, nullptr);

    add_ln_kernel<<<BS / 8, 256, 0, stream>>>(
        h, sa, ln1_w + (size_t)l * DD, ln1_b + (size_t)l * DD, h);

    gemm_f16w<2><<<dim3(FFD / 128, BS / 128), 256, 0, stream>>>(
        h, ff1_w + (size_t)l * FFD * DD, ff1_b + (size_t)l * FFD,
        big, FFD, DD, nullptr);

    gemm_f16w<0><<<dim3(DD / 128, BS / 128), 256, 0, stream>>>(
        big, ff2_w + (size_t)l * DD * FFD, ff2_b + (size_t)l * DD,
        sa, DD, FFD, nullptr);

    add_ln_kernel<<<BS / 8, 256, 0, stream>>>(
        h, sa, ln2_w + (size_t)l * DD, ln2_b + (size_t)l * DD, h);
  }

  final_kernel<<<(BB * INDIM + 255) / 256, 256, 0, stream>>>(h, W_fc, b_fc, out);
}
